// TorchPQCodec_23725399343286
// MI455X (gfx1250) — matrix operand order repaired and checked by reading, not compiled, not run on hardware
//
#include <hip/hip_runtime.h>
#include <hip/hip_bf16.h>

typedef __attribute__((ext_vector_type(16))) _Float16 v16h;
typedef __attribute__((ext_vector_type(8)))  float    v8f;

// Problem constants (fixed by setup_inputs)
#define N_Q   4096
#define M_T   16384
#define D_DIM 512
#define M_SUB 64
#define KSUB  256
#define DSUB  8

// GEMM tiling: 128x256 block, BK=32, 8 waves, each wave 64x64 (4x4 WMMA tiles)
#define BM 128
#define BN 256
#define BK 32
#define LROW 40  // padded LDS row length in halfs (80B) to spread banks

// ---------------------------------------------------------------------------
// Kernel 1: PQ encode queries -> write reconstructed query rows in f16.
// One block = (one subspace s) x (256 queries); codebook staged in LDS.
// ---------------------------------------------------------------------------
__global__ __launch_bounds__(256) void pq_encode(const float* __restrict__ x,
                                                 const float* __restrict__ cent,
                                                 _Float16* __restrict__ Qh) {
  const int s = blockIdx.y;
  const int i = blockIdx.x * 256 + threadIdx.x;
  __shared__ float cs[KSUB * DSUB];
  __shared__ float nrm[KSUB];
  {
    const int k = threadIdx.x;
    const float* c = cent + (size_t)(s * KSUB + k) * DSUB;
    float nn = 0.f;
#pragma unroll
    for (int d = 0; d < DSUB; ++d) {
      float v = c[d];
      cs[k * DSUB + d] = v;
      nn += v * v;
    }
    nrm[k] = nn;
  }
  __syncthreads();

  float xv[DSUB];
  const float* xp = x + (size_t)i * D_DIM + s * DSUB;
#pragma unroll
  for (int d = 0; d < DSUB; ++d) xv[d] = xp[d];

  float bd = 3.4e38f;
  int bk = 0;
  for (int k = 0; k < KSUB; ++k) {  // LDS broadcast reads: all lanes same addr
    float dot = 0.f;
#pragma unroll
    for (int d = 0; d < DSUB; ++d) dot = fmaf(xv[d], cs[k * DSUB + d], dot);
    float dis = nrm[k] - 2.0f * dot;
    if (dis < bd) { bd = dis; bk = k; }  // strict '<' == argmin first-occurrence
  }

  _Float16* q = Qh + (size_t)i * D_DIM + s * DSUB;
#pragma unroll
  for (int d = 0; d < DSUB; ++d) q[d] = (_Float16)cs[bk * DSUB + d];
}

// ---------------------------------------------------------------------------
// Kernel 2: decode target codes -> reconstructed datastore rows in f16.
// One thread per (row j, subspace s).
// ---------------------------------------------------------------------------
__global__ __launch_bounds__(256) void pq_decode(const int* __restrict__ codes,
                                                 const float* __restrict__ cent,
                                                 _Float16* __restrict__ Dh) {
  const int t = blockIdx.x * 256 + threadIdx.x;  // t = j*M_SUB + s
  const int j = t >> 6;
  const int s = t & 63;
  const int k = codes[t];
  const float* c = cent + (size_t)(s * KSUB + k) * DSUB;
  _Float16* d = Dh + (size_t)j * D_DIM + s * DSUB;
#pragma unroll
  for (int dd = 0; dd < DSUB; ++dd) d[dd] = (_Float16)c[dd];
}

// ---------------------------------------------------------------------------
// Kernel 3: sim = Q (4096x512) * D^T (512x16384), f16 in, f32 accumulate.
// ---------------------------------------------------------------------------
__device__ __forceinline__ v16h frag2(const _Float16* p0, const _Float16* p1) {
  union { uint4 u[2]; v16h h; } u;
  u.u[0] = *reinterpret_cast<const uint4*>(p0);
  u.u[1] = *reinterpret_cast<const uint4*>(p1);
  return u.h;
}

__global__ __launch_bounds__(256) void pq_gemm(const _Float16* __restrict__ Qh,
                                               const _Float16* __restrict__ Dh,
                                               float* __restrict__ out) {
  __shared__ __align__(16) _Float16 As[BM * LROW];  // 10.0 KB
  __shared__ __align__(16) _Float16 Bs[BN * LROW];  // 20.0 KB

  const int tid   = threadIdx.x;
  const int lane  = tid & 31;
  const int wave  = tid >> 5;
  const int waveR = wave & 1;   // 2 row-waves * 64 rows
  const int waveC = wave >> 1;  // 4 col-waves * 64 cols
  const int l     = lane & 15;
  const int hi    = lane >> 4;

  const int rowBase = blockIdx.y * BM;
  const int colBase = blockIdx.x * BN;

  // LDS fill mapping: 32B chunks. A: 256 chunks (1/thread). B: 512 (2/thread).
  const int arow = tid >> 1;
  const int aseg = (tid & 1) * 16;  // halfs

  const uint4* gA  = (const uint4*)(Qh + (size_t)(rowBase + arow) * D_DIM + aseg);
  const uint4* gB0 = (const uint4*)(Dh + (size_t)(colBase + arow) * D_DIM + aseg);
  const uint4* gB1 = (const uint4*)(Dh + (size_t)(colBase + 128 + arow) * D_DIM + aseg);

  uint4* lA  = (uint4*)(As + arow * LROW + aseg);
  uint4* lB0 = (uint4*)(Bs + arow * LROW + aseg);
  uint4* lB1 = (uint4*)(Bs + (128 + arow) * LROW + aseg);

  // Software pipeline: prefetch first tile into registers.
  uint4 pa0 = gA[0], pa1 = gA[1];
  uint4 pb00 = gB0[0], pb01 = gB0[1];
  uint4 pb10 = gB1[0], pb11 = gB1[1];

  v8f acc[4][4] = {};

  for (int k0 = 0; k0 < D_DIM; k0 += BK) {
    lA[0] = pa0;  lA[1] = pa1;
    lB0[0] = pb00; lB0[1] = pb01;
    lB1[0] = pb10; lB1[1] = pb11;
    __syncthreads();

    if (k0 + BK < D_DIM) {  // prefetch next tile; overlaps WMMA below
      gA += 4; gB0 += 4; gB1 += 4;  // advance BK halfs = 64B
      pa0 = gA[0];  pa1 = gA[1];
      pb00 = gB0[0]; pb01 = gB0[1];
      pb10 = gB1[0]; pb11 = gB1[1];
    }

    // A 16-bit 16x32 ISA layout: lane<16 -> K{0..7,16..23}; lane>=16 -> K{8..15,24..31}
    v16h af[4];
#pragma unroll
    for (int tm = 0; tm < 4; ++tm) {
      const _Float16* base = As + (waveR * 64 + tm * 16 + l) * LROW;
      af[tm] = frag2(base + hi * 8, base + 16 + hi * 8);
    }
    // B 16-bit 32x16 layout: lane = column, lane<16 -> K{0..7,16..23}; lane>=16 -> K{8..15,24..31}
#pragma unroll
    for (int tn = 0; tn < 4; ++tn) {
      const _Float16* bb = Bs + (waveC * 64 + tn * 16 + l) * LROW + hi * 8;
      v16h bf = frag2(bb, bb + 16);
#pragma unroll
      for (int tm = 0; tm < 4; ++tm)
        acc[tm][tn] = __builtin_amdgcn_wmma_f32_16x16x32_f16(
            false, af[tm], false, bf, (short)0, acc[tm][tn], false, false);
    }
    __syncthreads();
  }

  // C/D layout: VGPR r -> M = r + 8*(lane>=16); N = lane%16
#pragma unroll
  for (int tm = 0; tm < 4; ++tm) {
    const int r0 = rowBase + waveR * 64 + tm * 16 + hi * 8;
#pragma unroll
    for (int tn = 0; tn < 4; ++tn) {
      const int c = colBase + waveC * 64 + tn * 16 + l;
#pragma unroll
      for (int r = 0; r < 8; ++r)
        out[(size_t)(r0 + r) * M_T + c] = acc[tm][tn][r];
    }
  }
}

// ---------------------------------------------------------------------------
extern "C" void kernel_launch(void* const* d_in, const int* in_sizes, int n_in,
                              void* d_out, int out_size, void* d_ws, size_t ws_size,
                              hipStream_t stream) {
  const float* x    = (const float*)d_in[0];   // [4096, 512]
  const float* cent = (const float*)d_in[1];   // [64, 256, 8]
  const int*   tgt  = (const int*)d_in[2];     // [16384, 64]
  float*       out  = (float*)d_out;           // [4096, 16384]

  _Float16* Qh = (_Float16*)d_ws;                       // 4 MB
  _Float16* Dh = Qh + (size_t)N_Q * D_DIM;              // 16 MB

  pq_encode<<<dim3(N_Q / 256, M_SUB), 256, 0, stream>>>(x, cent, Qh);
  pq_decode<<<dim3((M_T * M_SUB) / 256), 256, 0, stream>>>(tgt, cent, Dh);
  pq_gemm<<<dim3(M_T / BN, N_Q / BM), 256, 0, stream>>>(Qh, Dh, out);
}
